// RESpecificTransformer_25486335934523
// MI455X (gfx1250) — compile-verified, matrix operand order checked by reading, not run on hardware
//
#include <hip/hip_runtime.h>

// ---------------------------------------------------------------------------
// MI455X / gfx1250 implementation of RESpecificTransformer reference.
// Dense math: v_wmma_f32_16x16x32_f16 (wave32 WMMA).
// A-tile staging: Tensor Data Mover (tensor_load_to_lds) + s_wait_tensorcnt.
// ---------------------------------------------------------------------------

#define USE_TDM 1

typedef _Float16 v16h __attribute__((ext_vector_type(16)));
typedef _Float16 v8h  __attribute__((ext_vector_type(8)));
typedef _Float16 v4h  __attribute__((ext_vector_type(4)));
typedef float    v8f  __attribute__((ext_vector_type(8)));
typedef float    v4f  __attribute__((ext_vector_type(4)));
typedef unsigned int v4u __attribute__((ext_vector_type(4)));
typedef int      v8i  __attribute__((ext_vector_type(8)));
typedef int      v4i  __attribute__((ext_vector_type(4)));

#define B_  8
#define T_  1024
#define E_  512
#define NH_ 8
#define HD_ 64
#define LBL_ 64
#define D_  576        // E + LBL
#define FF1_ 256
#define R_  16
#define NPAIR_ 276     // C(24,2)
#define MROWS_ 8192    // B*T
#define PROWS_ 2208    // B*NPAIR

__device__ inline v8f zero_v8f() {
  v8f v;
#pragma unroll
  for (int i = 0; i < 8; ++i) v[i] = 0.0f;
  return v;
}

// Build a 16-half WMMA fragment from two contiguous 8-half (16-byte) chunks.
// Matches the CDNA5 16-bit A/B VGPR layout: halves[0..7] = K g*8..g*8+7,
// halves[8..15] = K 16+g*8..16+g*8+7 for lane group g = lane/16.
__device__ inline v16h make_frag(const _Float16* lo, const _Float16* hi) {
  v8h a = *reinterpret_cast<const v8h*>(lo);
  v8h b = *reinterpret_cast<const v8h*>(hi);
  v16h r;
#pragma unroll
  for (int i = 0; i < 8; ++i) { r[i] = a[i]; r[i + 8] = b[i]; }
  return r;
}

__device__ inline v8f wmma_f16(v16h a, v16h b, v8f c) {
  return __builtin_amdgcn_wmma_f32_16x16x32_f16(
      /*neg_a=*/false, a, /*neg_b=*/false, b,
      /*c_mod=*/(short)0, c, /*reuse_a=*/false, /*reuse_b=*/false);
}

#define BM 64
#define BN 128
#define BK 32

#if USE_TDM
// Issue one TDM descriptor: load a BM x BK tile of f16 elements from a
// row-major [rows x >=cols] tensor (row stride = stride_elems) into LDS at
// byte offset lds_off. OOB rows/cols (beyond tensor_dim) read as zero.
// D# layout per CDNA5 ISA ch.8 (group0 128b, group1 256b, groups2/3 zero).
__device__ inline void tdm_load_tile_f16(const _Float16* gaddr, unsigned lds_off,
                                         int rows, int cols, int stride_elems)
{
  unsigned long long ga = (unsigned long long)(uintptr_t)gaddr;
  v4u g0;
  g0[0] = 1u;                                            // count=1 (valid user D#)
  g0[1] = lds_off;                                       // lds_addr (bytes)
  g0[2] = (unsigned)(ga & 0xFFFFFFFFu);                  // global_addr[31:0]
  g0[3] = (unsigned)((ga >> 32) & 0x01FFFFFFu) | (2u << 30); // [120:64] addr, type=2
  v8i g1;
  unsigned td0 = (unsigned)cols;                         // tensor_dim0 (elements)
  unsigned td1 = (unsigned)rows;                         // tensor_dim1 (rows)
  g1[0] = (int)(1u << 16);                               // wg_mask=0, data_size=1 (2B)
  g1[1] = (int)((td0 & 0xFFFFu) << 16);                  // tensor_dim0[15:0] @ [63:48]
  g1[2] = (int)(((td0 >> 16) & 0xFFFFu) | ((td1 & 0xFFFFu) << 16));
  g1[3] = (int)(((td1 >> 16) & 0xFFFFu) | ((unsigned)BK << 16)); // tile_dim0=BK
  g1[4] = (int)((unsigned)BM);                           // tile_dim1=BM, tile_dim2=0
  g1[5] = (int)(unsigned)stride_elems;                   // tensor_dim0_stride[31:0]
  g1[6] = 0;                                             // stride[47:32], dim1_stride lo
  g1[7] = 0;                                             // dim1_stride hi
  v4i gz; gz[0] = 0; gz[1] = 0; gz[2] = 0; gz[3] = 0;
#if __clang_major__ >= 23
  v8i gz8;
#pragma unroll
  for (int i = 0; i < 8; ++i) gz8[i] = 0;
  __builtin_amdgcn_tensor_load_to_lds(g0, g1, gz, gz, gz8, 0);
#else
  __builtin_amdgcn_tensor_load_to_lds(g0, g1, gz, gz, 0);
#endif
}
#endif

// ---------------------------------------------------------------------------
// Generic GEMM: C[M,N] = act(A16[M,K] @ W[N,K]^T + bias)   (einsum 'mk,nk->mn')
// A is f16 (lda = K stride), W is f32 (row n contiguous, ldw = K stride).
// Block tile 64(M) x 128(N), 8 waves, each wave 32x32 via 4 WMMAs per K-chunk.
// ---------------------------------------------------------------------------
__global__ __launch_bounds__(256)
void gemm_wmma_kernel(const _Float16* __restrict__ A, int lda,
                      const float* __restrict__ W, int ldw,
                      const float* __restrict__ bias,
                      float* __restrict__ Cf, _Float16* __restrict__ Ch, int ldc,
                      int M, int N, int K, int relu)
{
  __shared__ _Float16 As[BM][BK];   // [m][k]
  __shared__ _Float16 Bs[BN][BK];   // [n][k]  (rows of W, f32->f16)

  const int tid  = threadIdx.x;
  const int wave = tid >> 5;
  const int lane = tid & 31;
  const int wm   = wave >> 2;       // 0..1
  const int wn   = wave & 3;        // 0..3
  const int g    = lane >> 4;
  const int ln   = lane & 15;
  const int m0   = blockIdx.y * BM;
  const int n0   = blockIdx.x * BN;

  v8f acc[2][2];
#pragma unroll
  for (int i = 0; i < 2; ++i)
#pragma unroll
    for (int j = 0; j < 2; ++j) acc[i][j] = zero_v8f();

  for (int kc = 0; kc < K; kc += BK) {
#if USE_TDM
    // ---- A tile via Tensor Data Mover (wave 0 issues the descriptor) ----
    if (wave == 0) {
      unsigned lds_off = (unsigned)(uintptr_t)(&As[0][0]);
      tdm_load_tile_f16(A + (size_t)m0 * lda + kc, lds_off,
                        M - m0, K - kc, lda);
    }
#else
    { // ---- A tile: 64 rows x 32 halves, one b128 per thread ----
      int m  = tid >> 2;
      int kk = (tid & 3) * 8;
      int gm = m0 + m; if (gm > M - 1) gm = M - 1;
      *reinterpret_cast<v8h*>(&As[m][kk]) =
          *reinterpret_cast<const v8h*>(A + (size_t)gm * lda + kc + kk);
    }
#endif
    // ---- B tile: 128 rows x 32 f32; batch 4 b128 loads, then cvt+store ----
    {
      v4f wv[4];
#pragma unroll
      for (int r = 0; r < 4; ++r) {
        int idx = r * 256 + tid;           // 0..1023 float4 chunks
        int n   = idx >> 3;                // 0..127
        int kk  = (idx & 7) * 4;
        int gn  = n0 + n; if (gn > N - 1) gn = N - 1;
        wv[r] = *reinterpret_cast<const v4f*>(W + (size_t)gn * ldw + kc + kk);
      }
#pragma unroll
      for (int r = 0; r < 4; ++r) {
        int idx = r * 256 + tid;
        int n   = idx >> 3;
        int kk  = (idx & 7) * 4;
        v4h h;
#pragma unroll
        for (int q = 0; q < 4; ++q) h[q] = (_Float16)wv[r][q];
        *reinterpret_cast<v4h*>(&Bs[n][kk]) = h;
      }
    }
    if (kc + BK < K) {   // warm next K-chunk (global_prefetch_b8)
      __builtin_prefetch(W + (size_t)(n0 + (tid >> 1)) * ldw + kc + BK, 0, 1);
    }
#if USE_TDM
    __builtin_amdgcn_s_wait_tensorcnt(0);  // TDM tile landed in LDS
#endif
    __syncthreads();

    v16h af[2], bf[2];
#pragma unroll
    for (int s = 0; s < 2; ++s) {
      int row = wm * 32 + s * 16 + ln;
      af[s] = make_frag(&As[row][g * 8], &As[row][16 + g * 8]);
    }
#pragma unroll
    for (int s = 0; s < 2; ++s) {
      int row = wn * 32 + s * 16 + ln;
      bf[s] = make_frag(&Bs[row][g * 8], &Bs[row][16 + g * 8]);
    }
#pragma unroll
    for (int i = 0; i < 2; ++i)
#pragma unroll
      for (int j = 0; j < 2; ++j)
        acc[i][j] = wmma_f16(af[i], bf[j], acc[i][j]);
    __syncthreads();
  }

  // ---- store: C layout lane holds n = ln, rows m = e + 8*g ----
#pragma unroll
  for (int i = 0; i < 2; ++i) {
#pragma unroll
    for (int j = 0; j < 2; ++j) {
      int gn = n0 + wn * 32 + j * 16 + ln;
      if (gn < N) {
        float bv = bias ? bias[gn] : 0.0f;
#pragma unroll
        for (int e = 0; e < 8; ++e) {
          int gm = m0 + wm * 32 + i * 16 + e + 8 * g;
          if (gm < M) {
            float v = acc[i][j][e] + bv;
            if (relu) v = fmaxf(v, 0.0f);
            if (Cf) Cf[(size_t)gm * ldc + gn] = v;
            if (Ch) Ch[(size_t)gm * ldc + gn] = (_Float16)v;
          }
        }
      }
    }
  }
}

// ---------------------------------------------------------------------------
// Flash attention: grid (T/128, NH, B); block = 256 thr = 8 waves.
// Each wave owns 16 query rows; K/V chunks of 32 keys staged in LDS.
// qkv layout: [B*T, 3E] f16, q|k|v at col offsets 0|512|1024, head h at +h*64.
// ---------------------------------------------------------------------------
__global__ __launch_bounds__(256)
void flash_attn_kernel(const _Float16* __restrict__ qkv,
                       _Float16* __restrict__ O, int T)
{
  __shared__ _Float16 Ks[32][HD_];        // [k][d]
  __shared__ _Float16 Vs[HD_][32];        // [d][k]  (transposed)
  __shared__ _Float16 Ps[8][16][32];      // per-wave P tile [m][k]

  const int tid  = threadIdx.x;
  const int wave = tid >> 5;
  const int lane = tid & 31;
  const int g    = lane >> 4;
  const int ln   = lane & 15;
  const int b    = blockIdx.z;
  const int h    = blockIdx.y;
  const int q0   = blockIdx.x * 128;

  const _Float16* Qb = qkv + (size_t)b * T * 1536 + h * HD_;
  const _Float16* Kb = Qb + 512;
  const _Float16* Vb = Qb + 1024;

  const int qrow = q0 + wave * 16 + ln;
  v16h qf[2];
#pragma unroll
  for (int c = 0; c < 2; ++c) {
    const _Float16* qp = Qb + (size_t)qrow * 1536 + c * 32;
    qf[c] = make_frag(qp + g * 8, qp + 16 + g * 8);
  }

  float mrow[8], srow[8];
  v8f o[4];
#pragma unroll
  for (int e = 0; e < 8; ++e) { mrow[e] = -1e30f; srow[e] = 0.0f; }
#pragma unroll
  for (int j = 0; j < 4; ++j) o[j] = zero_v8f();

  const float sc = 0.125f;   // 1/sqrt(64)
  const int kr = tid >> 3;
  const int dd = (tid & 7) * 8;

  for (int kc = 0; kc < T; kc += 32) {
    // stage K chunk: 32x64 halves, one b128 per thread
    v8h kv = *reinterpret_cast<const v8h*>(Kb + (size_t)(kc + kr) * 1536 + dd);
    // stage V chunk transposed: one b128 load, 8 b16 LDS stores per thread
    v8h vv = *reinterpret_cast<const v8h*>(Vb + (size_t)(kc + kr) * 1536 + dd);
    *reinterpret_cast<v8h*>(&Ks[kr][dd]) = kv;
#pragma unroll
    for (int q = 0; q < 8; ++q) Vs[dd + q][kr] = vv[q];
    __syncthreads();

    // S = Q @ K^T for 32 keys -> two 16x16 tiles
    v8f s0 = zero_v8f(), s1 = zero_v8f();
#pragma unroll
    for (int c = 0; c < 2; ++c) {
      v16h k0 = make_frag(&Ks[ln][c * 32 + g * 8],      &Ks[ln][c * 32 + 16 + g * 8]);
      v16h k1 = make_frag(&Ks[16 + ln][c * 32 + g * 8], &Ks[16 + ln][c * 32 + 16 + g * 8]);
      s0 = wmma_f16(qf[c], k0, s0);
      s1 = wmma_f16(qf[c], k1, s1);
    }

    // online softmax (row m = e + 8*g; cols k = ln and 16+ln)
    float p0[8], p1[8];
#pragma unroll
    for (int e = 0; e < 8; ++e) {
      float v0 = s0[e] * sc, v1 = s1[e] * sc;
      float cm = fmaxf(v0, v1);
#pragma unroll
      for (int m = 1; m < 16; m <<= 1) cm = fmaxf(cm, __shfl_xor(cm, m, 32));
      float nm = fmaxf(mrow[e], cm);
      float corr = __expf(mrow[e] - nm);
      mrow[e] = nm;
      p0[e] = __expf(v0 - nm);
      p1[e] = __expf(v1 - nm);
      float rs = p0[e] + p1[e];
#pragma unroll
      for (int m = 1; m < 16; m <<= 1) rs += __shfl_xor(rs, m, 32);
      srow[e] = srow[e] * corr + rs;
#pragma unroll
      for (int j = 0; j < 4; ++j) o[j][e] *= corr;
    }

    // write P (C-layout regs -> [m][k] in LDS)
#pragma unroll
    for (int e = 0; e < 8; ++e) {
      Ps[wave][8 * g + e][ln]      = (_Float16)p0[e];
      Ps[wave][8 * g + e][16 + ln] = (_Float16)p1[e];
    }
    __syncthreads();

    // O += P @ V  (A = P re-read in A-layout, B = V^T rows contiguous in k)
    v16h pf = make_frag(&Ps[wave][ln][g * 8], &Ps[wave][ln][16 + g * 8]);
#pragma unroll
    for (int j = 0; j < 4; ++j) {
      v16h vf = make_frag(&Vs[j * 16 + ln][g * 8], &Vs[j * 16 + ln][16 + g * 8]);
      o[j] = wmma_f16(pf, vf, o[j]);
    }
    __syncthreads();
  }

  // normalize + write (rows m = e + 8*g)
#pragma unroll
  for (int j = 0; j < 4; ++j) {
#pragma unroll
    for (int e = 0; e < 8; ++e) {
      int grow = q0 + wave * 16 + 8 * g + e;
      int col  = h * HD_ + j * 16 + ln;
      float den = fmaxf(srow[e], 1e-20f);
      O[((size_t)b * T + grow) * E_ + col] = (_Float16)(o[j][e] / den);
    }
  }
}

// ---------------------------------------------------------------------------
// Elementwise / reduction helper kernels
// ---------------------------------------------------------------------------
__global__ void init_kernel(const float* __restrict__ src,
                            float* __restrict__ Xf, _Float16* __restrict__ Xh)
{
  size_t i = (size_t)blockIdx.x * 256 + threadIdx.x;
  if (i < (size_t)MROWS_ * E_) {
    float v = src[i];
    Xf[i] = v;
    Xh[i] = (_Float16)v;
  }
}

__global__ __launch_bounds__(256)
void ln_residual_kernel(const float* __restrict__ X, const float* __restrict__ F,
                        const float* __restrict__ gam, const float* __restrict__ bet,
                        float* __restrict__ Xout, _Float16* __restrict__ Xh)
{
  const int row = blockIdx.x;
  const int tid = threadIdx.x;
  const float* xr = X + (size_t)row * E_;
  const float* fr = F + (size_t)row * E_;
  float a0 = xr[tid] + fr[tid];
  float a1 = xr[tid + 256] + fr[tid + 256];
  float s = a0 + a1, s2 = a0 * a0 + a1 * a1;
#pragma unroll
  for (int m = 1; m < 32; m <<= 1) { s += __shfl_xor(s, m, 32); s2 += __shfl_xor(s2, m, 32); }
  __shared__ float ws[8], ws2[8];
  int wave = tid >> 5;
  if ((tid & 31) == 0) { ws[wave] = s; ws2[wave] = s2; }
  __syncthreads();
  if (tid == 0) {
    float t = 0, t2 = 0;
    for (int w = 0; w < 8; ++w) { t += ws[w]; t2 += ws2[w]; }
    ws[0] = t; ws2[0] = t2;
  }
  __syncthreads();
  float mu = ws[0] * (1.0f / E_);
  float var = ws2[0] * (1.0f / E_) - mu * mu;
  float rstd = rsqrtf(var + 1e-5f);
  float y0 = (a0 - mu) * rstd * gam[tid] + bet[tid];
  float y1 = (a1 - mu) * rstd * gam[tid + 256] + bet[tid + 256];
  Xout[(size_t)row * E_ + tid] = y0;
  Xout[(size_t)row * E_ + tid + 256] = y1;
  Xh[(size_t)row * E_ + tid] = (_Float16)y0;
  Xh[(size_t)row * E_ + tid + 256] = (_Float16)y1;
}

__global__ __launch_bounds__(256)
void bn_stats_kernel(const float* __restrict__ X, float* __restrict__ mu,
                     float* __restrict__ rstd)
{
  const int e = blockIdx.x;
  const int tid = threadIdx.x;
  float s = 0, s2 = 0;
  for (int r = tid; r < MROWS_; r += 256) {
    float v = X[(size_t)r * E_ + e];
    s += v; s2 += v * v;
  }
#pragma unroll
  for (int m = 1; m < 32; m <<= 1) { s += __shfl_xor(s, m, 32); s2 += __shfl_xor(s2, m, 32); }
  __shared__ float ws[8], ws2[8];
  int wave = tid >> 5;
  if ((tid & 31) == 0) { ws[wave] = s; ws2[wave] = s2; }
  __syncthreads();
  if (tid == 0) {
    float t = 0, t2 = 0;
    for (int w = 0; w < 8; ++w) { t += ws[w]; t2 += ws2[w]; }
    float m0 = t / (float)MROWS_;
    float v0 = t2 / (float)MROWS_ - m0 * m0;
    mu[e] = m0;
    rstd[e] = rsqrtf(v0 + 1e-5f);
  }
}

__global__ void gather_emb_kernel(const float* __restrict__ shr, const float* __restrict__ X,
                                  const float* __restrict__ ner, const int* __restrict__ rs,
                                  const int* __restrict__ re, const float* __restrict__ mu,
                                  const float* __restrict__ rstd,
                                  float* __restrict__ EMB, _Float16* __restrict__ EMBh)
{
  int idx = blockIdx.x * 256 + threadIdx.x;
  if (idx >= B_ * 24 * D_) return;
  int d = idx % D_;
  int j = (idx / D_) % 24;
  int b = idx / (D_ * 24);
  int tok = (j < 12) ? rs[b * 12 + j] : re[b * 12 + j - 12];
  float v;
  if (d < E_) {
    size_t o = ((size_t)b * T_ + tok) * E_ + d;
    v = shr[o] + (X[o] - mu[d]) * rstd[d];
  } else {
    v = ner[((size_t)b * T_ + tok) * LBL_ + (d - E_)];
  }
  EMB[idx] = v;
  EMBh[idx] = (_Float16)v;
}

__device__ inline void pair_ij(int p, int& i, int& j) {
  i = 0;
  for (;;) {
    int c = 23 - i;
    if (p < c) { j = i + 1 + p; return; }
    p -= c; ++i;
  }
}

__global__ void pair_build_kernel(const _Float16* __restrict__ EMBh,
                                  _Float16* __restrict__ Aa, _Float16* __restrict__ Ab)
{
  int idx = blockIdx.x * 256 + threadIdx.x;
  if (idx >= PROWS_ * D_) return;
  int d = idx % D_;
  int row = idx / D_;
  int b = row / NPAIR_, p = row % NPAIR_;
  int i, j; pair_ij(p, i, j);
  Aa[idx] = EMBh[((size_t)b * 24 + i) * D_ + d];
  Ab[idx] = EMBh[((size_t)b * 24 + j) * D_ + d];
}

__global__ void zero_kernel(float* out) { out[0] = 0.0f; }

__device__ inline float red64(float v, float* sh) {
#pragma unroll
  for (int m = 1; m < 32; m <<= 1) v += __shfl_xor(v, m, 32);
  int wave = threadIdx.x >> 5;
  if ((threadIdx.x & 31) == 0) sh[wave] = v;
  __syncthreads();
  float r = sh[0] + sh[1];
  __syncthreads();
  return r;
}

__global__ __launch_bounds__(64)
void score_bce_kernel(const float* __restrict__ H1, const float* __restrict__ H2,
                      const float* __restrict__ EMB, const float* __restrict__ Mten,
                      const float* __restrict__ W2, const float* __restrict__ b2,
                      const int* __restrict__ rs, const int* __restrict__ re,
                      const int* __restrict__ rq, float* __restrict__ out)
{
  __shared__ float sh[2];
  __shared__ float shbce[16];
  const int row = blockIdx.x;
  const int b = row / NPAIR_, p = row % NPAIR_;
  int i, j; pair_ij(p, i, j);
  const float* h1 = H1 + (size_t)row * FF1_;
  const float* h2 = H2 + (size_t)row * FF1_;
  const float* av = EMB + ((size_t)b * 24 + i) * D_;
  const float* bv = EMB + ((size_t)b * 24 + j) * D_;
  const int t = threadIdx.x;

  float p1 = 0, p2 = 0, pd = 0;
  for (int d = t; d < FF1_; d += 64) {
    float x = h1[d], y = h2[d];
    p1 += x * x; p2 += y * y; pd += x * y;
  }
  float dmp[R_];
#pragma unroll
  for (int r = 0; r < R_; ++r) dmp[r] = 0.0f;
  for (int d = t; d < D_; d += 64) {
    float pr = av[d] * bv[d];
#pragma unroll
    for (int r = 0; r < R_; ++r)
      dmp[r] += pr * Mten[(size_t)r * D_ * D_ + (size_t)d * (D_ + 1)];
  }
  float n1sq = red64(p1, sh);
  float n2sq = red64(p2, sh);
  float dot  = red64(pd, sh);
  float dm[R_];
  for (int r = 0; r < R_; ++r) dm[r] = red64(dmp[r], sh);

  float cosv = dot / (fmaxf(sqrtf(n1sq), 1e-8f) * fmaxf(sqrtf(n2sq), 1e-8f));

  int pf = (i < 12) ? rs[b * 12 + i] : re[b * 12 + i - 12];
  int ps = (j < 12) ? rs[b * 12 + j] : re[b * 12 + j - 12];

  if (t < R_) {
    int r = t;
    const float* wr = W2 + (size_t)r * (2 * FF1_ + 1 + R_);
    float acc = b2[r];
    for (int d = 0; d < FF1_; ++d) acc += h1[d] * wr[d];
    for (int d = 0; d < FF1_; ++d) acc += h2[d] * wr[FF1_ + d];
    acc += cosv * wr[2 * FF1_];
    for (int k = 0; k < R_; ++k) acc += dm[k] * wr[2 * FF1_ + 1 + k];
    float s = 1.0f / (1.0f + expf(-acc));
    float tgt = 0.0f;
    for (int k = 0; k < 12; ++k)
      if (pf == rs[b * 12 + k] && ps == re[b * 12 + k] && rq[b * 12 + k] == r) tgt = 1.0f;
    float logp   = fmaxf(logf(s), -100.0f);
    float log1mp = fmaxf(log1pf(-s), -100.0f);
    shbce[r] = -(tgt * logp + (1.0f - tgt) * log1mp);
  }
  __syncthreads();
  if (t == 0) {
    float sum = 0.0f;
    for (int r = 0; r < R_; ++r) sum += shbce[r];
    atomicAdd(out, sum * (1.0f / R_));
  }
}

// ---------------------------------------------------------------------------
// Launch sequence
// ---------------------------------------------------------------------------
extern "C" void kernel_launch(void* const* d_in, const int* in_sizes, int n_in,
                              void* d_out, int out_size, void* d_ws, size_t ws_size,
                              hipStream_t stream)
{
  (void)in_sizes; (void)n_in; (void)out_size; (void)ws_size;
  const float* x0   = (const float*)d_in[0];
  const float* ner  = (const float*)d_in[1];
  const int*   rst  = (const int*)d_in[2];
  const int*   ren  = (const int*)d_in[3];
  const int*   rsq  = (const int*)d_in[4];
  const float* wqkv = (const float*)d_in[5];
  const float* bqkv = (const float*)d_in[6];
  const float* wo   = (const float*)d_in[7];
  const float* bo   = (const float*)d_in[8];
  const float* g1   = (const float*)d_in[9];
  const float* be1  = (const float*)d_in[10];
  const float* g2   = (const float*)d_in[11];
  const float* be2  = (const float*)d_in[12];
  const float* w1   = (const float*)d_in[13];
  const float* bb1  = (const float*)d_in[14];
  const float* w2   = (const float*)d_in[15];
  const float* bb2  = (const float*)d_in[16];
  const float* f1w  = (const float*)d_in[17];
  const float* f1b  = (const float*)d_in[18];
  const float* f2w  = (const float*)d_in[19];
  const float* f2b  = (const float*)d_in[20];
  const float* Mten = (const float*)d_in[21];
  float* out = (float*)d_out;

  char* ws = (char*)d_ws;
  size_t off = 0;
  auto alloc = [&](size_t bytes) -> char* {
    char* pp = ws + off;
    off += (bytes + 255) & ~(size_t)255;
    return pp;
  };
  float*    Xf   = (float*)alloc((size_t)MROWS_ * E_ * 4);
  float*    Tf   = (float*)alloc((size_t)MROWS_ * E_ * 4);
  _Float16* X16  = (_Float16*)alloc((size_t)MROWS_ * E_ * 2);
  _Float16* QKV  = (_Float16*)alloc((size_t)MROWS_ * 3 * E_ * 2);
  _Float16* O16  = (_Float16*)alloc((size_t)MROWS_ * E_ * 2);
  _Float16* H16  = (_Float16*)alloc((size_t)MROWS_ * E_ * 2);
  float*    mu   = (float*)alloc(E_ * 4);
  float*    rstd = (float*)alloc(E_ * 4);
  float*    EMB  = (float*)alloc((size_t)B_ * 24 * D_ * 4);
  _Float16* EMBh = (_Float16*)alloc((size_t)B_ * 24 * D_ * 2);
  _Float16* Aa   = (_Float16*)alloc((size_t)PROWS_ * D_ * 2);
  _Float16* Ab   = (_Float16*)alloc((size_t)PROWS_ * D_ * 2);
  float*    H1   = (float*)alloc((size_t)PROWS_ * FF1_ * 4);
  float*    H2   = (float*)alloc((size_t)PROWS_ * FF1_ * 4);

  dim3 blk(256);
  init_kernel<<<(MROWS_ * E_ + 255) / 256, blk, 0, stream>>>(x0, Xf, X16);

  for (int l = 0; l < 2; ++l) {
    gemm_wmma_kernel<<<dim3(3 * E_ / BN, MROWS_ / BM), blk, 0, stream>>>(
        X16, E_, wqkv + (size_t)l * 3 * E_ * E_, E_, bqkv + (size_t)l * 3 * E_,
        nullptr, QKV, 3 * E_, MROWS_, 3 * E_, E_, 0);
    flash_attn_kernel<<<dim3(T_ / 128, NH_, B_), blk, 0, stream>>>(QKV, O16, T_);
    gemm_wmma_kernel<<<dim3(E_ / BN, MROWS_ / BM), blk, 0, stream>>>(
        O16, E_, wo + (size_t)l * E_ * E_, E_, bo + (size_t)l * E_,
        Tf, nullptr, E_, MROWS_, E_, E_, 0);
    ln_residual_kernel<<<MROWS_, blk, 0, stream>>>(Xf, Tf, g1 + l * E_, be1 + l * E_, Xf, X16);
    gemm_wmma_kernel<<<dim3(E_ / BN, MROWS_ / BM), blk, 0, stream>>>(
        X16, E_, w1 + (size_t)l * E_ * E_, E_, bb1 + (size_t)l * E_,
        nullptr, H16, E_, MROWS_, E_, E_, 1);
    gemm_wmma_kernel<<<dim3(E_ / BN, MROWS_ / BM), blk, 0, stream>>>(
        H16, E_, w2 + (size_t)l * E_ * E_, E_, bb2 + (size_t)l * E_,
        Tf, nullptr, E_, MROWS_, E_, E_, 0);
    ln_residual_kernel<<<MROWS_, blk, 0, stream>>>(Xf, Tf, g2 + l * E_, be2 + l * E_, Xf, X16);
  }

  bn_stats_kernel<<<E_, blk, 0, stream>>>(Xf, mu, rstd);
  gather_emb_kernel<<<(B_ * 24 * D_ + 255) / 256, blk, 0, stream>>>(
      x0, Xf, ner, rst, ren, mu, rstd, EMB, EMBh);
  pair_build_kernel<<<(PROWS_ * D_ + 255) / 256, blk, 0, stream>>>(EMBh, Aa, Ab);

  gemm_wmma_kernel<<<dim3((FF1_ + BN - 1) / BN, (PROWS_ + BM - 1) / BM), blk, 0, stream>>>(
      Aa, D_, f1w, D_, f1b, H1, nullptr, FF1_, PROWS_, FF1_, D_, 1);
  gemm_wmma_kernel<<<dim3((FF1_ + BN - 1) / BN, (PROWS_ + BM - 1) / BM), blk, 0, stream>>>(
      Ab, D_, f1w, D_, f1b, H2, nullptr, FF1_, PROWS_, FF1_, D_, 1);

  zero_kernel<<<1, 1, 0, stream>>>(out);
  score_bce_kernel<<<PROWS_, dim3(64), 0, stream>>>(
      H1, H2, EMB, Mten, f2w, f2b, rst, ren, rsq, out);
}
